// QuantLinear_85349590106242
// MI455X (gfx1250) — 8-bit matrix operand order repaired and checked by reading, compiled, not run on hardware
//
#include <hip/hip_runtime.h>
#include <hip/hip_bf16.h>

#define TOKENS 4096
#define IN_F   4096
#define OUT_F  11008
#define KPACK  (IN_F / 2)

typedef int   v2i __attribute__((ext_vector_type(2)));
typedef int   v4i __attribute__((ext_vector_type(4)));
typedef int   v8i __attribute__((ext_vector_type(8)));

__device__ __forceinline__ void async_b128(unsigned ldsAddr, const void* gptr)
{
    asm volatile("global_load_async_to_lds_b128 %0, %1, off"
                 :: "v"(ldsAddr), "v"(gptr) : "memory");
}
template <int N>
__device__ __forceinline__ void wait_async()
{
    asm volatile("s_wait_asynccnt %0" :: "i"(N) : "memory");
}

__device__ __forceinline__ int pack2(int p0, int p1)
{
    const int l0 = ((p0 & 15) ^ 8) - 8;
    const int h0 = (((p0 >> 4) & 15) ^ 8) - 8;
    const int l1 = ((p1 & 15) ^ 8) - 8;
    const int h1 = (((p1 >> 4) & 15) ^ 8) - 8;
    return (int)((unsigned)(l0 & 0xFF)        | ((unsigned)(h0 & 0xFF) << 8) |
                ((unsigned)(l1 & 0xFF) << 16) | ((unsigned)(h1 & 0xFF) << 24));
}

__global__ __launch_bounds__(256)
void quant_act_kernel(const float* __restrict__ x,
                      signed char* __restrict__ xq,
                      float* __restrict__ scales)
{
    __shared__ float red[256];
    const int row = blockIdx.x;
    const int t   = threadIdx.x;
    const float* xr = x + (size_t)row * IN_F;

    float4 v[4];
    float m = 0.0f;
#pragma unroll
    for (int g = 0; g < 4; ++g) {
        v[g] = ((const float4*)xr)[g * 256 + t];
        m = fmaxf(m, fmaxf(fmaxf(fabsf(v[g].x), fabsf(v[g].y)),
                           fmaxf(fabsf(v[g].z), fabsf(v[g].w))));
    }
    red[t] = m;
    __syncthreads();
#pragma unroll
    for (int s = 128; s > 0; s >>= 1) {
        if (t < s) red[t] = fmaxf(red[t], red[t + s]);
        __syncthreads();
    }
    const float mx  = red[0];
    const float inv = (mx > 0.0f) ? (127.0f / mx) : 0.0f;
    if (t == 0) scales[row] = mx / 127.0f;

    int* oq = (int*)(xq + (size_t)row * IN_F);
#pragma unroll
    for (int g = 0; g < 4; ++g) {
        int q0 = __float2int_rn(v[g].x * inv);
        int q1 = __float2int_rn(v[g].y * inv);
        int q2 = __float2int_rn(v[g].z * inv);
        int q3 = __float2int_rn(v[g].w * inv);
        q0 = min(127, max(-127, q0));
        q1 = min(127, max(-127, q1));
        q2 = min(127, max(-127, q2));
        q3 = min(127, max(-127, q3));
        oq[g * 256 + t] = (q0 & 0xFF) | ((q1 & 0xFF) << 8) |
                          ((q2 & 0xFF) << 16) | ((q3 & 0xFF) << 24);
    }
}

__global__ __launch_bounds__(256)
void unpack_w_kernel(const int* __restrict__ wp, signed char* __restrict__ w8)
{
    const size_t gid = (size_t)blockIdx.x * 256 + threadIdx.x;
    const v4i* src = (const v4i*)wp + gid * 2;
    const v4i p0 = src[0];
    const v4i p1 = src[1];
    v4i u;
    u.x = pack2(p0.x, p0.y);
    u.y = pack2(p0.z, p0.w);
    u.z = pack2(p1.x, p1.y);
    u.w = pack2(p1.z, p1.w);
    ((v4i*)w8)[gid] = u;
}

#define A_ST 80
#define B_ST 80
#define A_BYTES (128 * A_ST)
#define B_BYTES (128 * B_ST)
#define TILE_BYTES (A_BYTES + B_BYTES)

__global__ __launch_bounds__(256)
void w8a8_gemm_async_kernel(const signed char* __restrict__ xq,
                            const float* __restrict__ ascale,
                            const signed char* __restrict__ w8,
                            const float* __restrict__ wscale,
                            const float* __restrict__ bias,
                            float* __restrict__ out)
{
    __shared__ __align__(16) unsigned char smem[3][TILE_BYTES];

    const int tid  = threadIdx.x;
    const int lane = tid & 31;
    const int wave = tid >> 5;
    const int m0   = blockIdx.y * 128;
    const int n0   = blockIdx.x * 128;

    const int mw = (wave & 3) * 32;
    const int nw = (wave >> 2) * 64;

    const int am = tid >> 1;
    const int ak = (tid & 1) * 32;
    const signed char* aGlob = xq + (size_t)(m0 + am) * IN_F + ak;
    const signed char* wGlob = w8 + (size_t)(n0 + am) * IN_F + ak;

    v8i acc[2][4] = {};

    auto asyncTile = [&](int kt, int buf) {
        unsigned char* s  = smem[buf];
        unsigned char* bB = s + A_BYTES;
        const int kb = kt * 64;
        async_b128((unsigned)(size_t)(s  + am * A_ST + ak),      aGlob + kb);
        async_b128((unsigned)(size_t)(s  + am * A_ST + ak + 16), aGlob + kb + 16);
        async_b128((unsigned)(size_t)(bB + am * B_ST + ak),      wGlob + kb);
        async_b128((unsigned)(size_t)(bB + am * B_ST + ak + 16), wGlob + kb + 16);
    };

    auto compute = [&](int buf) {
        const unsigned char* s  = smem[buf];
        const unsigned char* bB = s + A_BYTES;
        const int ml = lane & 15;
        const int kA = (lane >> 4) * 16;
        const int kB = (lane >> 4) * 16;
        v8i af[2];
#pragma unroll
        for (int fm = 0; fm < 2; ++fm) {
            const unsigned char* ap = s + (mw + fm * 16 + ml) * A_ST;
            v2i p0 = *(const v2i*)(ap + kA);
            v2i p1 = *(const v2i*)(ap + kA + 8);
            v2i p2 = *(const v2i*)(ap + kA + 32);
            v2i p3 = *(const v2i*)(ap + kA + 40);
            af[fm][0] = p0.x; af[fm][1] = p0.y;
            af[fm][2] = p1.x; af[fm][3] = p1.y;
            af[fm][4] = p2.x; af[fm][5] = p2.y;
            af[fm][6] = p3.x; af[fm][7] = p3.y;
        }
#pragma unroll
        for (int fn = 0; fn < 4; ++fn) {
            const unsigned char* bp = bB + (nw + fn * 16 + ml) * B_ST;
            v4i q0 = *(const v4i*)(bp + kB);
            v4i q1 = *(const v4i*)(bp + 32 + kB);
            v8i bf;
            bf[0] = q0.x; bf[1] = q0.y; bf[2] = q0.z; bf[3] = q0.w;
            bf[4] = q1.x; bf[5] = q1.y; bf[6] = q1.z; bf[7] = q1.w;
#pragma unroll
            for (int fm = 0; fm < 2; ++fm)
                acc[fm][fn] = __builtin_amdgcn_wmma_i32_16x16x64_iu8(
                     true, af[fm],  true, bf,
                    acc[fm][fn],  false,  false);
        }
    };

    auto step = [&](int kt, int curbuf, int nextTileBuf) {
        asyncTile(kt + 2, nextTileBuf);
        compute(curbuf);
        wait_async<4>();
        __syncthreads();
    };

    asyncTile(0, 0);
    asyncTile(1, 1);
    wait_async<4>();
    __syncthreads();

    for (int kt = 0; kt < 60; kt += 3) {
        step(kt + 0, 0, 2);
        step(kt + 1, 1, 0);
        step(kt + 2, 2, 1);
    }
    step(60, 0, 2);
    step(61, 1, 0);
    compute(2);
    wait_async<0>();
    __syncthreads();
    compute(0);

#pragma unroll
    for (int fm = 0; fm < 2; ++fm) {
        const int trow = m0 + mw + fm * 16 + ((lane >> 4) * 8);
        float as[8];
#pragma unroll
        for (int r = 0; r < 8; ++r) as[r] = ascale[trow + r];
#pragma unroll
        for (int fn = 0; fn < 4; ++fn) {
            const int n = n0 + nw + fn * 16 + (lane & 15);
            const float wsc = wscale[n];
            const float bv  = bias[n];
#pragma unroll
            for (int r = 0; r < 8; ++r) {
                out[(size_t)(trow + r) * OUT_F + n] =
                    (float)acc[fm][fn][r] * (as[r] * wsc) + bv;
            }
        }
    }
}

#define FB_A_BYTES (128 * A_ST)
#define FB_B_BYTES (64 * B_ST)

__global__ __launch_bounds__(256)
void w4a8_gemm_fused_kernel(const signed char* __restrict__ xq,
                            const float* __restrict__ ascale,
                            const int* __restrict__ wpacked,
                            const float* __restrict__ wscale,
                            const float* __restrict__ bias,
                            float* __restrict__ out)
{
    __shared__ __align__(16) unsigned char smem[2][FB_A_BYTES + FB_B_BYTES];

    const int tid  = threadIdx.x;
    const int lane = tid & 31;
    const int wave = tid >> 5;
    const int m0   = blockIdx.y * 128;
    const int n0   = blockIdx.x * 64;

    const int mw = (wave & 3) * 32;
    const int nw = (wave >> 2) * 32;

    const int am = tid >> 1;
    const int ak = (tid & 1) * 32;
    const int bn = tid >> 2;
    const int bk = (tid & 3) * 16;

    const signed char* aGlob = xq + (size_t)(m0 + am) * IN_F + ak;
    const int*         wGlob = wpacked + (size_t)(n0 + bn) * KPACK + (bk >> 1);

    v8i acc[2][2] = {};
    v4i ra0, ra1, rw0, rw1;

    auto loadStage = [&](int kt) {
        const int kb = kt * 64;
        const v4i* ap = (const v4i*)(aGlob + kb);
        ra0 = ap[0]; ra1 = ap[1];
        const v4i* wp = (const v4i*)(wGlob + (kb >> 1));
        rw0 = wp[0]; rw1 = wp[1];
    };
    auto storeStage = [&](int buf) {
        unsigned char* s = smem[buf];
        *(v4i*)(s + am * A_ST + ak)      = ra0;
        *(v4i*)(s + am * A_ST + ak + 16) = ra1;
        unsigned char* bB = s + FB_A_BYTES;
        v4i u;
        u.x = pack2(rw0.x, rw0.y); u.y = pack2(rw0.z, rw0.w);
        u.z = pack2(rw1.x, rw1.y); u.w = pack2(rw1.z, rw1.w);
        *(v4i*)(bB + bn * B_ST + bk) = u;
    };
    auto compute = [&](int buf) {
        const unsigned char* s  = smem[buf];
        const unsigned char* bB = s + FB_A_BYTES;
        const int ml = lane & 15;
        const int kA = (lane >> 4) * 16;
        const int kB = (lane >> 4) * 16;
        v8i af[2], bf[2];
#pragma unroll
        for (int f = 0; f < 2; ++f) {
            const unsigned char* ap = s + (mw + f * 16 + ml) * A_ST;
            v2i p0 = *(const v2i*)(ap + kA);
            v2i p1 = *(const v2i*)(ap + kA + 8);
            v2i p2 = *(const v2i*)(ap + kA + 32);
            v2i p3 = *(const v2i*)(ap + kA + 40);
            af[f][0] = p0.x; af[f][1] = p0.y;
            af[f][2] = p1.x; af[f][3] = p1.y;
            af[f][4] = p2.x; af[f][5] = p2.y;
            af[f][6] = p3.x; af[f][7] = p3.y;
            const unsigned char* bp = bB + (nw + f * 16 + ml) * B_ST;
            v4i q0 = *(const v4i*)(bp + kB);
            v4i q1 = *(const v4i*)(bp + 32 + kB);
            bf[f][0] = q0.x; bf[f][1] = q0.y; bf[f][2] = q0.z; bf[f][3] = q0.w;
            bf[f][4] = q1.x; bf[f][5] = q1.y; bf[f][6] = q1.z; bf[f][7] = q1.w;
        }
#pragma unroll
        for (int fm = 0; fm < 2; ++fm)
#pragma unroll
            for (int fn = 0; fn < 2; ++fn)
                acc[fm][fn] = __builtin_amdgcn_wmma_i32_16x16x64_iu8(
                    true, af[fm], true, bf[fn], acc[fm][fn], false, false);
    };

    loadStage(0);
    storeStage(0);
    __syncthreads();

    for (int kt = 0; kt < 62; kt += 2) {
        loadStage(kt + 1); compute(0); storeStage(1); __syncthreads();
        loadStage(kt + 2); compute(1); storeStage(0); __syncthreads();
    }
    loadStage(63); compute(0); storeStage(1); __syncthreads();
    compute(1);

#pragma unroll
    for (int fm = 0; fm < 2; ++fm) {
        const int trow = m0 + mw + fm * 16 + ((lane >> 4) * 8);
        float as[8];
#pragma unroll
        for (int r = 0; r < 8; ++r) as[r] = ascale[trow + r];
#pragma unroll
        for (int fn = 0; fn < 2; ++fn) {
            const int n = n0 + nw + fn * 16 + (lane & 15);
            const float wsc = wscale[n];
            const float bv  = bias[n];
#pragma unroll
            for (int r = 0; r < 8; ++r) {
                out[(size_t)(trow + r) * OUT_F + n] =
                    (float)acc[fm][fn][r] * (as[r] * wsc) + bv;
            }
        }
    }
}

extern "C" void kernel_launch(void* const* d_in, const int* in_sizes, int n_in,
                              void* d_out, int out_size, void* d_ws, size_t ws_size,
                              hipStream_t stream)
{
    (void)in_sizes; (void)n_in; (void)out_size;

    const float* x    = (const float*)d_in[0];
    const int*   w    = (const int*)  d_in[1];
    const float* wsc  = (const float*)d_in[2];
    const float* bias = (const float*)d_in[3];
    float*       out  = (float*)d_out;

    const size_t xqBytes  = (size_t)TOKENS * IN_F;
    const size_t scOff    = xqBytes;
    const size_t w8Off    = xqBytes + 65536;
    const size_t w8Bytes  = (size_t)OUT_F * IN_F;
    const size_t needFast = w8Off + w8Bytes;

    signed char* xq     = (signed char*)d_ws;
    float*       ascale = (float*)((char*)d_ws + scOff);

    quant_act_kernel<<<TOKENS, 256, 0, stream>>>(x, xq, ascale);

    if (ws_size >= needFast) {
        signed char* w8 = (signed char*)((char*)d_ws + w8Off);
        unpack_w_kernel<<<OUT_F, 256, 0, stream>>>(w, w8);
        dim3 grid(OUT_F / 128, TOKENS / 128);
        w8a8_gemm_async_kernel<<<grid, 256, 0, stream>>>(xq, ascale, w8, wsc, bias, out);
    } else {
        dim3 grid(OUT_F / 64, TOKENS / 128);
        w4a8_gemm_fused_kernel<<<grid, 256, 0, stream>>>(xq, ascale, w, wsc, bias, out);
    }
}
